// VCRVQAGNNModel_44178033607057
// MI455X (gfx1250) — compile-verified, matrix operand order checked by reading, not run on hardware
//
#include <hip/hip_runtime.h>
#include <hip/hip_bf16.h>

// ---------------------------------------------------------------------------
// CDNA5 (gfx1250) implementation. wave32, WMMA 16x16x32 bf16 -> f32,
// TDM (tensor_load_to_lds) weight-panel staging with double buffering.
// ---------------------------------------------------------------------------

typedef __attribute__((ext_vector_type(16))) __bf16 v16bf;
typedef __attribute__((ext_vector_type(8)))  __bf16 v8bf;
typedef __attribute__((ext_vector_type(4)))  __bf16 v4bf;
typedef __attribute__((ext_vector_type(8)))  float  v8f;
typedef __attribute__((ext_vector_type(4)))  unsigned int v4u;
typedef __attribute__((ext_vector_type(8)))  int v8i;
typedef __attribute__((ext_vector_type(4)))  int v4i;

#define DH   1024
#define NT   1024
#define Bsz  8
#define NV_  100
#define NKG_ 923
#define Lq   128
#define Hh   8
#define DHD_ 128

#if defined(__gfx1250__) && __has_builtin(__builtin_amdgcn_tensor_load_to_lds) && \
    __has_builtin(__builtin_amdgcn_s_wait_tensorcnt)
#define HAVE_TDM 1
#else
#define HAVE_TDM 0
#endif

// Load one half-row fragment: 8 contiguous bf16 at base, 8 more at base+16.
// Matches CDNA5 16-bit A/B operand layout (per-lane K runs).
static __device__ inline v16bf load_frag(const __bf16* base) {
  v8bf lo = *(const v8bf*)(base);
  v8bf hi = *(const v8bf*)(base + 16);
  return __builtin_shufflevector(lo, hi, 0,1,2,3,4,5,6,7,8,9,10,11,12,13,14,15);
}

#if HAVE_TDM
// Issue a 2D TDM tile load: tile_d0 x tile_d1 elements of 2 bytes from a
// row-major tensor (row stride = stride0 elements) into LDS at lds_addr.
// D# layout per CDNA5 ISA 8.3/8.4 (group0: count/lds/global/type,
// group1: data_size, dims, tile dims, strides).
static __device__ inline void tdm_load_2d(unsigned lds_addr, const void* gptr,
                                          unsigned tensor_d0, unsigned tensor_d1,
                                          unsigned tile_d0, unsigned tile_d1,
                                          unsigned long long stride0)
{
  unsigned long long ga = (unsigned long long)gptr;
  v4u g0;
  g0.x = 1u;                                                  // count=1, user D#
  g0.y = lds_addr;                                            // LDS byte address
  g0.z = (unsigned)ga;                                        // global addr lo
  g0.w = (unsigned)((ga >> 32) & 0x01FFFFFFu) | (2u << 30);   // addr hi | type=2
  v8i g1;
  g1[0] = (int)(1u << 16);                                    // data_size=1 (2B)
  g1[1] = (int)((tensor_d0 & 0xFFFFu) << 16);                 // tensor_dim0 lo16
  g1[2] = (int)(((tensor_d0 >> 16) & 0xFFFFu) |
                ((tensor_d1 & 0xFFFFu) << 16));               // d0 hi | d1 lo
  g1[3] = (int)(((tensor_d1 >> 16) & 0xFFFFu) |
                ((tile_d0 & 0xFFFFu) << 16));                 // d1 hi | tile_d0
  g1[4] = (int)(tile_d1 & 0xFFFFu);                           // tile_d1, tile_d2=0
  g1[5] = (int)(unsigned)(stride0 & 0xFFFFFFFFull);           // dim0 stride lo32
  g1[6] = (int)(unsigned)((stride0 >> 32) & 0xFFFFull);       // dim0 stride hi16
  g1[7] = 0;
  v4i gz = {0, 0, 0, 0};
#if __clang_major__ >= 23
  v8i gz8 = {0, 0, 0, 0, 0, 0, 0, 0};
  __builtin_amdgcn_tensor_load_to_lds(g0, g1, gz, gz, gz8, 0);
#else
  __builtin_amdgcn_tensor_load_to_lds(g0, g1, gz, gz, 0);
#endif
}
#endif

// ---------------------------------------------------------------------------
// Blocked bf16 GEMM: C[Mp,N] = A[Mp,Kp] * W[Kp,N] + bias, W pre-transposed
// as Bt[N,Kp].  Block = 4 waves = 128M x 64N tile; each wave computes
// 32M x 64N (8 WMMAs / K-chunk).  The 64N x 32K weight panel is staged in
// LDS (TDM when available), double-buffered, shared by all 4 waves.
// ---------------------------------------------------------------------------
__global__ void __launch_bounds__(128) gemm_bf16_wmma(
    const __bf16* __restrict__ A, const __bf16* __restrict__ Bt,
    const float* __restrict__ bias, float* __restrict__ C,
    int Mp, int N, int Kp)
{
  __shared__ __bf16 bpanel[2][64 * 32];
  const int tid  = threadIdx.x;
  const int lane = tid & 31;
  const int wave = tid >> 5;
  const int lr   = lane & 15;
  const int half = lane >> 4;
  const int kb   = half * 8;
  const int m0   = blockIdx.y * 128 + wave * 32;
  const int n0   = blockIdx.x * 64;

  int mr0 = m0 + lr;       if (mr0 >= Mp) mr0 = Mp - 1;
  int mr1 = m0 + 16 + lr;  if (mr1 >= Mp) mr1 = Mp - 1;
  const __bf16* arow0 = A + (size_t)mr0 * Kp;
  const __bf16* arow1 = A + (size_t)mr1 * Kp;

  auto stage_load = [&](int p, int k0) {
#if HAVE_TDM
    if (wave == 0) {
      tdm_load_2d((unsigned)(size_t)(void*)&bpanel[p][0],
                  (const void*)(Bt + (size_t)n0 * Kp + k0),
                  (unsigned)Kp, (unsigned)N, 32u, 64u,
                  (unsigned long long)Kp);
    }
#else
    // cooperative global->LDS copy: 64 rows x 32 bf16
#pragma unroll
    for (int q = 0; q < 2; ++q) {
      const int idx = tid + q * 128;           // 256 8-elem segments
      const int row = idx >> 2;
      const int seg = (idx & 3) * 8;
      *(v8bf*)&bpanel[p][row * 32 + seg] =
          *(const v8bf*)(Bt + (size_t)(n0 + row) * Kp + k0 + seg);
    }
#endif
  };

  v8f acc[2][4];
#pragma unroll
  for (int ms = 0; ms < 2; ++ms)
#pragma unroll
    for (int ns = 0; ns < 4; ++ns)
#pragma unroll
      for (int v = 0; v < 8; ++v) acc[ms][ns][v] = 0.f;

  stage_load(0, 0);
#if HAVE_TDM
  if (wave == 0) __builtin_amdgcn_s_wait_tensorcnt(0);
#endif
  __syncthreads();

  int stage = 0;
  for (int k0 = 0; k0 < Kp; k0 += 32) {
    const int next = stage ^ 1;
    if (k0 + 32 < Kp) stage_load(next, k0 + 32);

    __builtin_prefetch(arow0 + k0 + 128, 0, 1);
    __builtin_prefetch(arow1 + k0 + 128, 0, 1);
    v16bf a0 = load_frag(arow0 + k0 + kb);
    v16bf a1 = load_frag(arow1 + k0 + kb);
#pragma unroll
    for (int ns = 0; ns < 4; ++ns) {
      v16bf bfr = load_frag(&bpanel[stage][(ns * 16 + lr) * 32 + kb]);
      acc[0][ns] = __builtin_amdgcn_wmma_f32_16x16x32_bf16(
          false, a0, false, bfr, (short)0, acc[0][ns], false, false);
      acc[1][ns] = __builtin_amdgcn_wmma_f32_16x16x32_bf16(
          false, a1, false, bfr, (short)0, acc[1][ns], false, false);
    }
#if HAVE_TDM
    if (wave == 0) __builtin_amdgcn_s_wait_tensorcnt(0);
#endif
    __syncthreads();
    stage = next;
  }

#pragma unroll
  for (int ms = 0; ms < 2; ++ms)
#pragma unroll
    for (int ns = 0; ns < 4; ++ns) {
      const int col = n0 + ns * 16 + lr;
      const float bv = bias ? bias[col] : 0.f;
#pragma unroll
      for (int v = 0; v < 8; ++v) {
        const int row = m0 + ms * 16 + v + half * 8;
        if (row < Mp) C[(size_t)row * N + col] = acc[ms][ns][v] + bv;
      }
    }
}

// ---------------------------------------------------------------------------
// Weight convert + transpose: out[N][Kp] = bf16(W[K][N]), zero-pad K->Kp.
// ---------------------------------------------------------------------------
__global__ void __launch_bounds__(256) convert_wt_kernel(
    const float* __restrict__ W, __bf16* __restrict__ out, int K, int N, int Kp)
{
  size_t idx = (size_t)blockIdx.x * 256 + threadIdx.x;
  size_t total = (size_t)N * Kp;
  if (idx >= total) return;
  int n = (int)(idx / Kp);
  int k = (int)(idx % Kp);
  out[idx] = (k < K) ? (__bf16)W[(size_t)k * N + n] : (__bf16)0.f;
}

// Activation convert: out[Mp][Kp] = bf16(in[M][K]), zero-padded.
__global__ void __launch_bounds__(256) convert_act_kernel(
    const float* __restrict__ in, __bf16* __restrict__ out, int M, int K, int Mp, int Kp)
{
  size_t idx = (size_t)blockIdx.x * 256 + threadIdx.x;
  size_t total = (size_t)Mp * Kp;
  if (idx >= total) return;
  int m = (int)(idx / Kp);
  int k = (int)(idx % Kp);
  out[idx] = (m < M && k < K) ? (__bf16)in[(size_t)m * K + k] : (__bf16)0.f;
}

// ---------------------------------------------------------------------------
// Token embedding + masked mean pool.  32 blocks (rows 8..31 zero pad).
// ---------------------------------------------------------------------------
__global__ void __launch_bounds__(256) embed_pool_kernel(
    const float* __restrict__ emb, const int* __restrict__ ids,
    const int* __restrict__ msk, float* __restrict__ pooled,
    __bf16* __restrict__ pooledb)
{
  const int tid = threadIdx.x;
  const int b = blockIdx.x;
  if (b >= Bsz) {
    for (int d = tid; d < DH; d += 256) {
      pooled[(size_t)b * DH + d] = 0.f;
      pooledb[(size_t)b * DH + d] = (__bf16)0.f;
    }
    return;
  }
  float cnt = 0.f;
  for (int l = 0; l < Lq; ++l) cnt += (float)msk[b * Lq + l];
  cnt = fmaxf(cnt, 1.f);
  for (int d = tid; d < DH; d += 256) {
    float acc = 0.f;
    for (int l = 0; l < Lq; ++l) {
      if (msk[b * Lq + l])
        acc += emb[(size_t)ids[b * Lq + l] * DH + d];
    }
    const float p = acc / cnt;
    pooled[(size_t)b * DH + d] = p;
    pooledb[(size_t)b * DH + d] = (__bf16)p;
  }
}

// ---------------------------------------------------------------------------
// Row LayerNorm (in place) with optional residual add and bf16 mirror.
// ---------------------------------------------------------------------------
__global__ void __launch_bounds__(256) row_ln_kernel(
    float* __restrict__ x, const float* __restrict__ resid,
    const float* __restrict__ g, const float* __restrict__ bta,
    __bf16* __restrict__ outbf)
{
  __shared__ float red[256];
  const int tid = threadIdx.x;
  const size_t base = (size_t)blockIdx.x * DH;
  float v[4];
  float sum = 0.f;
#pragma unroll
  for (int k = 0; k < 4; ++k) {
    const int d = tid + k * 256;
    float t = x[base + d];
    if (resid) t += resid[base + d];
    v[k] = t;
    sum += t;
  }
  red[tid] = sum; __syncthreads();
  for (int off = 128; off > 0; off >>= 1) {
    if (tid < off) red[tid] += red[tid + off];
    __syncthreads();
  }
  const float mean = red[0] * (1.f / DH);
  __syncthreads();
  float vs = 0.f;
#pragma unroll
  for (int k = 0; k < 4; ++k) { const float d0 = v[k] - mean; vs += d0 * d0; }
  red[tid] = vs; __syncthreads();
  for (int off = 128; off > 0; off >>= 1) {
    if (tid < off) red[tid] += red[tid + off];
    __syncthreads();
  }
  const float rstd = rsqrtf(red[0] * (1.f / DH) + 1e-5f);
  __syncthreads();
#pragma unroll
  for (int k = 0; k < 4; ++k) {
    const int d = tid + k * 256;
    const float o = (v[k] - mean) * rstd * g[d] + bta[d];
    x[base + d] = o;
    if (outbf) outbf[base + d] = (__bf16)o;
  }
}

// ---------------------------------------------------------------------------
// Assemble x = concat(vis, q, kg) + type_emb[types]; emit f32 + bf16.
// ---------------------------------------------------------------------------
__global__ void __launch_bounds__(256) assemble_x_kernel(
    const float* __restrict__ vis, const float* __restrict__ q,
    const float* __restrict__ kg, const float* __restrict__ type_emb,
    const int* __restrict__ types, float* __restrict__ x, __bf16* __restrict__ xb)
{
  const int row = blockIdx.x;          // b*NT + node
  const int b = row >> 10;
  const int node = row & (NT - 1);
  const float* src;
  if (node < NV_)        src = vis + (size_t)(b * NV_ + node) * DH;
  else if (node == NV_)  src = q + (size_t)b * DH;
  else                   src = kg + (size_t)(b * NKG_ + node - NV_ - 1) * DH;
  const float* te = type_emb + (size_t)types[row] * DH;
  for (int d = threadIdx.x; d < DH; d += 256) {
    const float v = src[d] + te[d];
    x[(size_t)row * DH + d] = v;
    xb[(size_t)row * DH + d] = (__bf16)v;
  }
}

// ---------------------------------------------------------------------------
// s/t per-head dot products + transposed bf16 copy hT[b][h][d][n].
// ---------------------------------------------------------------------------
__global__ void __launch_bounds__(128) compute_st_kernel(
    const float* __restrict__ h, const float* __restrict__ asrc,
    const float* __restrict__ adst, float* __restrict__ s,
    float* __restrict__ t, __bf16* __restrict__ hT)
{
  __shared__ float red[128];
  const int tid = threadIdx.x;         // d within head
  const int row = blockIdx.x;          // b*NT + n
  const int b = row >> 10;
  const int n = row & (NT - 1);
  for (int hd = 0; hd < Hh; ++hd) {
    const float v = h[(size_t)row * DH + hd * DHD_ + tid];
    hT[((size_t)(b * Hh + hd) * DHD_ + tid) * NT + n] = (__bf16)v;
    red[tid] = v * asrc[hd * DHD_ + tid]; __syncthreads();
    for (int off = 64; off > 0; off >>= 1) {
      if (tid < off) red[tid] += red[tid + off];
      __syncthreads();
    }
    if (tid == 0) s[(size_t)row * Hh + hd] = red[0];
    __syncthreads();
    red[tid] = v * adst[hd * DHD_ + tid]; __syncthreads();
    for (int off = 64; off > 0; off >>= 1) {
      if (tid < off) red[tid] += red[tid + off];
      __syncthreads();
    }
    if (tid == 0) t[(size_t)row * Hh + hd] = red[0];
    __syncthreads();
  }
}

// ---------------------------------------------------------------------------
// Streaming softmax stats: per (b,i,head) row max and 1/sum.
// ---------------------------------------------------------------------------
__global__ void __launch_bounds__(256) attn_stats_kernel(
    const float* __restrict__ s, const float* __restrict__ t,
    const int* __restrict__ adj, float* __restrict__ mOut, float* __restrict__ rzOut)
{
  __shared__ float red[256];
  __shared__ float mfin[Hh];
  const int tid = threadIdx.x;
  const int row = blockIdx.x;          // b*NT + i
  const int b = row >> 10;
  const int* mrow = adj + (size_t)row * NT;
  const float* tb = t + (size_t)b * NT * Hh;
  float si[Hh];
#pragma unroll
  for (int hd = 0; hd < Hh; ++hd) si[hd] = s[(size_t)row * Hh + hd];
  float mx[Hh];
#pragma unroll
  for (int hd = 0; hd < Hh; ++hd) mx[hd] = -1e30f;
  for (int j = tid; j < NT; j += 256) {
    const int mk = mrow[j];
#pragma unroll
    for (int hd = 0; hd < Hh; ++hd) {
      float e = si[hd] + tb[j * Hh + hd];
      e = e > 0.f ? e : 0.2f * e;
      e = mk > 0 ? e : -1e9f;
      mx[hd] = fmaxf(mx[hd], e);
    }
  }
  for (int hd = 0; hd < Hh; ++hd) {
    red[tid] = mx[hd]; __syncthreads();
    for (int off = 128; off > 0; off >>= 1) {
      if (tid < off) red[tid] = fmaxf(red[tid], red[tid + off]);
      __syncthreads();
    }
    if (tid == 0) mfin[hd] = red[0];
    __syncthreads();
  }
  float sm[Hh];
#pragma unroll
  for (int hd = 0; hd < Hh; ++hd) sm[hd] = 0.f;
  for (int j = tid; j < NT; j += 256) {
    const int mk = mrow[j];
#pragma unroll
    for (int hd = 0; hd < Hh; ++hd) {
      float e = si[hd] + tb[j * Hh + hd];
      e = e > 0.f ? e : 0.2f * e;
      e = mk > 0 ? e : -1e9f;
      sm[hd] += __expf(e - mfin[hd]);
    }
  }
  for (int hd = 0; hd < Hh; ++hd) {
    red[tid] = sm[hd]; __syncthreads();
    for (int off = 128; off > 0; off >>= 1) {
      if (tid < off) red[tid] += red[tid + off];
      __syncthreads();
    }
    if (tid == 0) {
      mOut[(size_t)row * Hh + hd] = mfin[hd];
      rzOut[(size_t)row * Hh + hd] = 1.f / red[0];
    }
    __syncthreads();
  }
}

// ---------------------------------------------------------------------------
// Attention aggregation via WMMA: block = 128 threads = 16i x 64d for one
// (b, head).  Each 16x32 bf16 probability tile is produced cooperatively
// into double-buffered LDS once, then shared by all 4 consumer waves.
// ---------------------------------------------------------------------------
__global__ void __launch_bounds__(128) attn_agg_kernel(
    const __bf16* __restrict__ hT, const float* __restrict__ s,
    const float* __restrict__ t, const float* __restrict__ mstat,
    const float* __restrict__ rzstat, const int* __restrict__ adj,
    float* __restrict__ outp)
{
  __shared__ __bf16 Pl[2][16 * 32];
  const int tid  = threadIdx.x;
  const int lane = tid & 31;
  const int wave = tid >> 5;
  const int itile = blockIdx.x >> 1;   // 64 i-tiles
  const int dgrp  = blockIdx.x & 1;    // 2 d-groups of 64
  const int head = blockIdx.y;
  const int b = blockIdx.z;
  const int half = lane >> 4;
  const int kb = half * 8;

  // producer role: row = tid/8 (0..15), cols pc0..pc0+3 of the 32-wide tile
  const int prow = tid >> 3;
  const int pc0  = (tid & 7) * 4;
  const size_t rowp = (size_t)(b * NT + itile * 16 + prow);
  const float sp  = s[rowp * Hh + head];
  const float mp  = mstat[rowp * Hh + head];
  const float rzp = rzstat[rowp * Hh + head];
  const int* mrowp = adj + rowp * NT;
  const float* tb = t + (size_t)b * NT * Hh + head;

  auto produce = [&](int p, int j0) {
    const int4 mk = *(const int4*)(mrowp + j0 + pc0);
    const int mv[4] = {mk.x, mk.y, mk.z, mk.w};
    v4bf pv;
#pragma unroll
    for (int q = 0; q < 4; ++q) {
      const int j = j0 + pc0 + q;
      float e = sp + tb[(size_t)j * Hh];
      e = e > 0.f ? e : 0.2f * e;
      e = mv[q] > 0 ? e : -1e9f;
      pv[q] = (__bf16)(__expf(e - mp) * rzp);
    }
    *(v4bf*)&Pl[p][prow * 32 + pc0] = pv;
  };

  // consumer role
  const int dloc = dgrp * 64 + wave * 16 + (lane & 15);   // 0..127 within head
  const __bf16* hrow = hT + ((size_t)(b * Hh + head) * DHD_ + dloc) * NT;
  v8f c = {0.f,0.f,0.f,0.f,0.f,0.f,0.f,0.f};

  produce(0, 0);
  __syncthreads();
  int stage = 0;
  for (int j0 = 0; j0 < NT; j0 += 32) {
    const int next = stage ^ 1;
    if (j0 + 32 < NT) produce(next, j0 + 32);
    v16bf a = load_frag(&Pl[stage][(lane & 15) * 32 + kb]);
    v16bf bv = load_frag(hrow + j0 + kb);
    c = __builtin_amdgcn_wmma_f32_16x16x32_bf16(false, a, false, bv,
                                                (short)0, c, false, false);
    __syncthreads();
    stage = next;
  }
#pragma unroll
  for (int v = 0; v < 8; ++v) {
    const int ir = itile * 16 + v + half * 8;
    outp[((size_t)(b * NT + ir)) * DH + head * DHD_ + dloc] = c[v];
  }
}

// ---------------------------------------------------------------------------
// Readout: scores, softmax-weighted pooling, final gelu scorer.
// ---------------------------------------------------------------------------
__global__ void __launch_bounds__(128) readout_scores_kernel(
    const float* __restrict__ x, const float* __restrict__ w,
    const float* __restrict__ bb, float* __restrict__ scores)
{
  __shared__ float red[128];
  const int tid = threadIdx.x;
  const size_t base = (size_t)blockIdx.x * DH;
  float acc = 0.f;
#pragma unroll
  for (int k = 0; k < 8; ++k) {
    const int d = tid + k * 128;
    acc += x[base + d] * w[d];
  }
  red[tid] = acc; __syncthreads();
  for (int off = 64; off > 0; off >>= 1) {
    if (tid < off) red[tid] += red[tid + off];
    __syncthreads();
  }
  if (tid == 0) scores[blockIdx.x] = red[0] + bb[0];
}

__global__ void __launch_bounds__(256) readout_pool_kernel(
    const float* __restrict__ x, const float* __restrict__ scores,
    float* __restrict__ g)
{
  __shared__ float red[256];
  __shared__ float w[NT];
  const int tid = threadIdx.x;
  const int b = blockIdx.x;
  const float* sc = scores + (size_t)b * NT;
  float mx = -1e30f;
  for (int n = tid; n < NT; n += 256) mx = fmaxf(mx, sc[n]);
  red[tid] = mx; __syncthreads();
  for (int off = 128; off > 0; off >>= 1) {
    if (tid < off) red[tid] = fmaxf(red[tid], red[tid + off]);
    __syncthreads();
  }
  const float m = red[0]; __syncthreads();
  float sm = 0.f;
  for (int n = tid; n < NT; n += 256) {
    const float e = __expf(sc[n] - m);
    w[n] = e;
    sm += e;
  }
  red[tid] = sm; __syncthreads();
  for (int off = 128; off > 0; off >>= 1) {
    if (tid < off) red[tid] += red[tid + off];
    __syncthreads();
  }
  const float rz = 1.f / red[0]; __syncthreads();
  for (int n = tid; n < NT; n += 256) w[n] *= rz;
  __syncthreads();
  for (int d = tid; d < DH; d += 256) {
    float acc = 0.f;
    const float* xb = x + (size_t)b * NT * DH + d;
    for (int n = 0; n < NT; ++n) acc += w[n] * xb[(size_t)n * DH];
    g[(size_t)b * DH + d] = acc;
  }
}

__global__ void __launch_bounds__(256) final_score_kernel(
    const float* __restrict__ u, const float* __restrict__ w2,
    const float* __restrict__ b2, float* __restrict__ out)
{
  __shared__ float red[256];
  const int tid = threadIdx.x;
  const int b = blockIdx.x;
  float acc = 0.f;
#pragma unroll
  for (int k = 0; k < 4; ++k) {
    const int d = tid + k * 256;
    const float xv = u[(size_t)b * DH + d];     // cs_b1 already added by GEMM
    const float gl = 0.5f * xv *
        (1.f + tanhf(0.7978845608f * (xv + 0.044715f * xv * xv * xv)));
    acc += gl * w2[d];
  }
  red[tid] = acc; __syncthreads();
  for (int off = 128; off > 0; off >>= 1) {
    if (tid < off) red[tid] += red[tid + off];
    __syncthreads();
  }
  if (tid == 0) out[b] = red[0] + b2[0];
}

// ---------------------------------------------------------------------------
// Host launcher
// ---------------------------------------------------------------------------
extern "C" void kernel_launch(void* const* d_in, const int* in_sizes, int n_in,
                              void* d_out, int out_size, void* d_ws, size_t ws_size,
                              hipStream_t stream) {
  (void)in_sizes; (void)n_in; (void)out_size; (void)ws_size;

  const float* visual   = (const float*)d_in[0];
  const int*   q_ids    = (const int*)d_in[1];
  const int*   q_mask   = (const int*)d_in[2];
  const float* kg_feat  = (const float*)d_in[3];
  const int*   adj      = (const int*)d_in[4];
  const int*   types    = (const int*)d_in[5];
  const float* vis_W    = (const float*)d_in[6];
  const float* vis_b    = (const float*)d_in[7];
  const float* vis_ln_g = (const float*)d_in[8];
  const float* vis_ln_b = (const float*)d_in[9];
  const float* tok_emb  = (const float*)d_in[10];
  const float* q_W      = (const float*)d_in[11];
  const float* q_b      = (const float*)d_in[12];
  const float* q_ln_g   = (const float*)d_in[13];
  const float* q_ln_b   = (const float*)d_in[14];
  const float* kg_W     = (const float*)d_in[15];
  const float* kg_b     = (const float*)d_in[16];
  const float* kg_ln_g  = (const float*)d_in[17];
  const float* kg_ln_b  = (const float*)d_in[18];
  const float* type_emb = (const float*)d_in[19];
  const float* gat_W    = (const float*)d_in[20];
  const float* gat_b    = (const float*)d_in[21];
  const float* a_src    = (const float*)d_in[22];
  const float* a_dst    = (const float*)d_in[23];
  const float* gat_ln_g = (const float*)d_in[24];
  const float* gat_ln_b = (const float*)d_in[25];
  const float* ro_W     = (const float*)d_in[26];
  const float* ro_b     = (const float*)d_in[27];
  const float* cs_W1    = (const float*)d_in[28];
  const float* cs_b1    = (const float*)d_in[29];
  const float* cs_W2    = (const float*)d_in[30];
  const float* cs_b2    = (const float*)d_in[31];
  float* out = (float*)d_out;

  // Workspace carve-out (256B aligned, deterministic layout).
  size_t off = 0;
  auto alloc = [&](size_t bytes) -> void* {
    void* p = (char*)d_ws + off;
    off += (bytes + 255) & ~(size_t)255;
    return p;
  };
  __bf16* wt_vis  = (__bf16*)alloc(1024ull * 2048 * 2);
  __bf16* wt_q    = (__bf16*)alloc(1024ull * 1024 * 2);
  __bf16* wt_kg   = (__bf16*)alloc(1024ull * 320 * 2);
  __bf16* wt_gat  = (__bf16*)alloc(3ull * 1024 * 1024 * 2);
  __bf16* wt_cs1  = (__bf16*)alloc(1024ull * 1024 * 2);
  __bf16* a_vis   = (__bf16*)alloc(800ull * 2048 * 2);
  __bf16* a_kg    = (__bf16*)alloc(7392ull * 320 * 2);
  float*  pooled  = (float*)alloc(32ull * 1024 * 4);
  __bf16* pooledb = (__bf16*)alloc(32ull * 1024 * 2);
  float*  c_vis   = (float*)alloc(800ull * 1024 * 4);
  float*  c_q     = (float*)alloc(32ull * 1024 * 4);
  float*  c_kg    = (float*)alloc(7392ull * 1024 * 4);
  float*  x       = (float*)alloc(8192ull * 1024 * 4);
  __bf16* xb      = (__bf16*)alloc(8192ull * 1024 * 2);
  float*  hbuf    = (float*)alloc(8192ull * 1024 * 4);
  __bf16* hT      = (__bf16*)alloc(8192ull * 1024 * 2);
  float*  sbuf    = (float*)alloc(8192ull * 8 * 4);
  float*  tbuf    = (float*)alloc(8192ull * 8 * 4);
  float*  mbuf    = (float*)alloc(8192ull * 8 * 4);
  float*  rzbuf   = (float*)alloc(8192ull * 8 * 4);
  float*  aout    = (float*)alloc(8192ull * 1024 * 4);
  float*  scores  = (float*)alloc(8192ull * 4);
  float*  gvec    = (float*)alloc(8ull * 1024 * 4);
  __bf16* gpadb   = (__bf16*)alloc(32ull * 1024 * 2);
  float*  ubuf    = (float*)alloc(32ull * 1024 * 4);

  // --- weight conversion / transposition (fp32 -> bf16 [N][Kp]) ---
  convert_wt_kernel<<<(1024 * 2048 + 255) / 256, 256, 0, stream>>>(vis_W, wt_vis, 2048, 1024, 2048);
  convert_wt_kernel<<<(1024 * 1024 + 255) / 256, 256, 0, stream>>>(q_W, wt_q, 1024, 1024, 1024);
  convert_wt_kernel<<<(1024 * 320 + 255) / 256, 256, 0, stream>>>(kg_W, wt_kg, 300, 1024, 320);
  for (int l = 0; l < 3; ++l)
    convert_wt_kernel<<<(1024 * 1024 + 255) / 256, 256, 0, stream>>>(
        gat_W + (size_t)l * 1024 * 1024, wt_gat + (size_t)l * 1024 * 1024, 1024, 1024, 1024);
  convert_wt_kernel<<<(1024 * 1024 + 255) / 256, 256, 0, stream>>>(cs_W1, wt_cs1, 1024, 1024, 1024);

  // --- activation conversion (zero-padded bf16) ---
  convert_act_kernel<<<(800 * 2048 + 255) / 256, 256, 0, stream>>>(visual, a_vis, 800, 2048, 800, 2048);
  convert_act_kernel<<<(7392 * 320 + 255) / 256, 256, 0, stream>>>(kg_feat, a_kg, 7384, 300, 7392, 320);

  // --- visual projection + LN ---
  gemm_bf16_wmma<<<dim3(16, 7), 128, 0, stream>>>(a_vis, wt_vis, vis_b, c_vis, 800, 1024, 2048);
  row_ln_kernel<<<800, 256, 0, stream>>>(c_vis, nullptr, vis_ln_g, vis_ln_b, nullptr);

  // --- KG projection + LN ---
  gemm_bf16_wmma<<<dim3(16, 58), 128, 0, stream>>>(a_kg, wt_kg, kg_b, c_kg, 7392, 1024, 320);
  row_ln_kernel<<<7384, 256, 0, stream>>>(c_kg, nullptr, kg_ln_g, kg_ln_b, nullptr);

  // --- text: embed + pool + proj + LN ---
  embed_pool_kernel<<<32, 256, 0, stream>>>(tok_emb, q_ids, q_mask, pooled, pooledb);
  gemm_bf16_wmma<<<dim3(16, 1), 128, 0, stream>>>(pooledb, wt_q, q_b, c_q, 32, 1024, 1024);
  row_ln_kernel<<<8, 256, 0, stream>>>(c_q, nullptr, q_ln_g, q_ln_b, nullptr);

  // --- assemble node tensor + type embeddings ---
  assemble_x_kernel<<<8192, 256, 0, stream>>>(c_vis, c_q, c_kg, type_emb, types, x, xb);

  // --- GAT stack ---
  for (int l = 0; l < 3; ++l) {
    gemm_bf16_wmma<<<dim3(16, 64), 128, 0, stream>>>(
        xb, wt_gat + (size_t)l * 1024 * 1024, gat_b + (size_t)l * 1024, hbuf, 8192, 1024, 1024);
    compute_st_kernel<<<8192, 128, 0, stream>>>(
        hbuf, a_src + (size_t)l * Hh * DHD_, a_dst + (size_t)l * Hh * DHD_, sbuf, tbuf, hT);
    attn_stats_kernel<<<8192, 256, 0, stream>>>(sbuf, tbuf, adj, mbuf, rzbuf);
    attn_agg_kernel<<<dim3(128, 8, 8), 128, 0, stream>>>(hT, sbuf, tbuf, mbuf, rzbuf, adj, aout);
    row_ln_kernel<<<8192, 256, 0, stream>>>(
        x, aout, gat_ln_g + (size_t)l * 1024, gat_ln_b + (size_t)l * 1024, xb);
  }

  // --- attention-pooled readout ---
  readout_scores_kernel<<<8192, 128, 0, stream>>>(x, ro_W, ro_b, scores);
  readout_pool_kernel<<<8, 256, 0, stream>>>(x, scores, gvec);

  // --- candidate scorer ---
  convert_act_kernel<<<(32 * 1024 + 255) / 256, 256, 0, stream>>>(gvec, gpadb, 8, 1024, 32, 1024);
  gemm_bf16_wmma<<<dim3(16, 1), 128, 0, stream>>>(gpadb, wt_cs1, cs_b1, ubuf, 32, 1024, 1024);
  final_score_kernel<<<8, 256, 0, stream>>>(ubuf, cs_W2, cs_b2, out);
}
